// MultiLayerRNN_32195074851539
// MI455X (gfx1250) — matrix operand order repaired and checked by reading, not compiled, not run on hardware
//
#include <hip/hip_runtime.h>
#include <hip/hip_bf16.h>
#include <math.h>

// ---------------- types ----------------
typedef __attribute__((ext_vector_type(16))) __bf16 v16bf;
typedef __attribute__((ext_vector_type(8)))  __bf16 v8bf;
typedef __attribute__((ext_vector_type(8)))  float  v8f;

#define S_LEN   512
#define BATCH   32
#define FDIM    1024
#define HDIM    1024
#define NLAYER  2

#define NWG_REC     32     // persistent workgroups for recurrence (each owns 32 H-columns)
#define REC_THREADS 256    // 8 waves (wave32)
// LDS: 3 gates * 32 rows * 1024 bf16 weights (192KB) + h slice f32 (4KB) + z slice f32 (4KB)
#define W_LDS_BYTES  (3*32*1024*2)
#define HSL_OFF      (W_LDS_BYTES)
#define ZSL_OFF      (W_LDS_BYTES + 32*32*4)
#define REC_LDS_BYTES (W_LDS_BYTES + 2*32*32*4)   // 204800 B (within 320KB WGP LDS)

// ---------------- helpers ----------------
__device__ __forceinline__ v16bf pack16(v8bf lo, v8bf hi) {
  v16bf r;
#pragma unroll
  for (int i = 0; i < 8; ++i) { r[i] = lo[i]; r[i + 8] = hi[i]; }
  return r;
}

__device__ __forceinline__ v8f wmma_bf16(v16bf a, v16bf b, v8f c) {
  // v_wmma_f32_16x16x32_bf16  (neg_a, A, neg_b, B, c_mod, C, reuse_a, reuse_b)
  return __builtin_amdgcn_wmma_f32_16x16x32_bf16(false, a, false, b, (short)0, c, false, false);
}

// Device-wide barrier: monotonic counter, agent-scope fences so cross-WGP data
// (h, r*h buffers) is visible past the WGP$ on the other side.
__device__ __forceinline__ void gbar(unsigned* cnt, unsigned* target) {
  __builtin_amdgcn_fence(__ATOMIC_RELEASE, "agent");
  __syncthreads();
  if (threadIdx.x == 0) {
    unsigned tgt = *target + gridDim.x;
    atomicAdd(cnt, 1u);
    while (__hip_atomic_load(cnt, __ATOMIC_RELAXED, __HIP_MEMORY_SCOPE_AGENT) < tgt) {
      __builtin_amdgcn_s_sleep(4);
    }
  }
  *target += gridDim.x;
  __syncthreads();
  __builtin_amdgcn_fence(__ATOMIC_ACQUIRE, "agent");
}

// ---------------- kernels ----------------
__global__ void gru_cvt_bf16(const float* __restrict__ src, __bf16* __restrict__ dst, size_t n) {
  size_t i = (size_t)blockIdx.x * blockDim.x + threadIdx.x;
  size_t stride = (size_t)gridDim.x * blockDim.x;
  for (; i < n; i += stride) dst[i] = (__bf16)src[i];
}

// Input projections: P_g[m][n] = sum_k X[m][k] * W_g[n][k] + bih_g[n]
// M = S*B = 16384, N = K = 1024.  grid = (8, 128, 3 gates), block = 256 (8 waves).
// Each wave: 64(M) x 32(N) strip = 8 accumulators (4 A-frags x 2 B-frags per k-step),
// ping-pong double-buffered fragments so WMMA overlaps the next k-block's loads.
__global__ void gru_proj_gemm(const __bf16* __restrict__ X,
                              const __bf16* __restrict__ Wr, const __bf16* __restrict__ Wz,
                              const __bf16* __restrict__ Wn,
                              const float* __restrict__ br, const float* __restrict__ bz,
                              const float* __restrict__ bn,
                              float* __restrict__ Pr, float* __restrict__ Pz,
                              float* __restrict__ Pn)
{
  const int K = 1024;
  const int g = blockIdx.z;
  const __bf16* W  = (g == 0) ? Wr : (g == 1) ? Wz : Wn;
  const float* bih = (g == 0) ? br : (g == 1) ? bz : bn;
  float* P         = (g == 0) ? Pr : (g == 1) ? Pz : Pn;

  const int tid  = threadIdx.x;
  const int wave = tid >> 5, lane = tid & 31;
  const int lr   = lane & 15, hi = lane >> 4;
  const int khiA = hi * 16, khiB = hi * 16;

  const int n0 = blockIdx.x * 128 + (wave & 3) * 32;   // 4 waves across N
  const int m0 = blockIdx.y * 128 + (wave >> 2) * 64;  // 2 waves across M

  const __bf16* arow = X + (size_t)(m0 + lr) * K;
  const __bf16* brow = W + (size_t)(n0 + lr) * K;

  v8f acc[4][2];
#pragma unroll
  for (int i = 0; i < 4; ++i)
#pragma unroll
    for (int j = 0; j < 2; ++j) acc[i][j] = v8f{0.f,0.f,0.f,0.f,0.f,0.f,0.f,0.f};

  struct FragSet { v8bf alo[4], ahi[4]; v16bf b[2]; };
  FragSet f0, f1;

  auto load_frags = [&](FragSet& f, int kb) {
#pragma unroll
    for (int i = 0; i < 4; ++i) {
      const __bf16* ap = arow + (size_t)i * 16 * K + kb * 32 + khiA;
      f.alo[i] = *(const v8bf*)ap;
      f.ahi[i] = *(const v8bf*)(ap + 8);
    }
#pragma unroll
    for (int j = 0; j < 2; ++j)
      f.b[j] = *(const v16bf*)(brow + (size_t)j * 16 * K + kb * 32 + khiB);
  };
  auto do_wmma = [&](FragSet& f) {
#pragma unroll
    for (int j = 0; j < 2; ++j) {
#pragma unroll
      for (int i = 0; i < 4; ++i)
        acc[i][j] = wmma_bf16(pack16(f.alo[i], f.ahi[i]), f.b[j], acc[i][j]);
    }
  };

  load_frags(f0, 0);
  for (int kb = 0; kb < 32; kb += 2) {
    load_frags(f1, kb + 1);                      // kb+1 <= 31
    do_wmma(f0);                                 // consumes kb
    load_frags(f0, (kb + 2 < 32) ? kb + 2 : 31); // last prefetch redundant, harmless
    do_wmma(f1);                                 // consumes kb+1
  }

  // epilogue: C/D layout -> VGPR v holds M = v (lanes<16) or v+8
  const int mrow = m0 + hi * 8;
#pragma unroll
  for (int j = 0; j < 2; ++j) {
    const int n = n0 + j * 16 + lr;
    const float bias = bih[n];
#pragma unroll
    for (int i = 0; i < 4; ++i) {
#pragma unroll
      for (int v = 0; v < 8; ++v)
        P[(size_t)(mrow + i * 16 + v) * HDIM + n] = acc[i][j][v] + bias;
    }
  }
}

// Persistent recurrence kernel. 32 WGs, each owns 32 columns of H.
// Whh slices (3 gates, bf16) live in LDS for all 512 steps.
// Per step: phase A (r,z GEMMs + r*h publish) -> grid barrier ->
//           phase B (n GEMM + h update + publish) -> grid barrier.
__global__ void gru_recurrent(const float* __restrict__ Pr, const float* __restrict__ Pz,
                              const float* __restrict__ Pn,
                              const float* __restrict__ bhr, const float* __restrict__ bhz,
                              const float* __restrict__ bhn,
                              const __bf16* __restrict__ Wr, const __bf16* __restrict__ Wz,
                              const __bf16* __restrict__ Wn,
                              __bf16* __restrict__ hbf,  __bf16* __restrict__ rhbf,
                              float* __restrict__ out_seq, __bf16* __restrict__ outbf_seq,
                              float* __restrict__ hn_out, unsigned* __restrict__ bar)
{
  extern __shared__ char smem[];
  __bf16* w_lds = (__bf16*)smem;                 // [3*32][1024] bf16
  float*  hsl   = (float*)(smem + HSL_OFF);      // [32][32] this WG's f32 h slice
  float*  zsl   = (float*)(smem + ZSL_OFF);      // [32][32] z gate staging

  const int K = HDIM;
  const int wg = blockIdx.x, tid = threadIdx.x;
  const int wave = tid >> 5, lane = tid & 31;
  const int lr = lane & 15, hi = lane >> 4;
  const int khiA = hi * 16, khiB = hi * 16;
  unsigned target = 0;

  // ---- preload Whh slices into LDS (192KB resident for whole sequence) ----
  for (int i = tid; i < 3 * 32 * 128; i += REC_THREADS) {   // 16B chunks
    int row = i >> 7;              // 0..95  (gate*32 + n_local)
    int c   = (i & 127) * 8;       // element offset within row
    int g = row >> 5, nl = row & 31;
    const __bf16* Wg = (g == 0) ? Wr : (g == 1) ? Wz : Wn;
    *(v8bf*)(w_lds + (size_t)row * 1024 + c) =
        *(const v8bf*)(Wg + ((size_t)(wg * 32 + nl)) * K + c);
  }
  // ---- h0 = 0 ----
  for (int i = tid; i < 32 * 32; i += REC_THREADS) {
    hsl[i] = 0.f; zsl[i] = 0.f;
    int m = i >> 5, nl = i & 31;
    hbf [(size_t)m * HDIM + wg * 32 + nl] = (__bf16)0.f;
    rhbf[(size_t)m * HDIM + wg * 32 + nl] = (__bf16)0.f;
  }
  gbar(bar, &target);

  // wave roles. Phase A: gate = wave>>2 (0=r,1=z), mtile=(wave>>1)&1, ntile=wave&1.
  // Phase B (waves 0..3): gate n, same (mtile,ntile) mapping.
  const int mtA = (wave >> 1) & 1, ntA = wave & 1, gA = wave >> 2;
  const int n_l = ntA * 16 + lr;
  const int n_g = wg * 32 + n_l;
  const __bf16* browA = w_lds + ((size_t)(gA * 32) + n_l) * 1024;
  const __bf16* browB = w_lds + ((size_t)(2 * 32) + n_l) * 1024;
  const float* PA  = gA ? Pz : Pr;
  const float* bhA = gA ? bhz : bhr;
  const __bf16* arowA = hbf  + (size_t)(mtA * 16 + lr) * HDIM;
  const __bf16* arowB = rhbf + (size_t)(mtA * 16 + lr) * HDIM;
  const int mbase = mtA * 16 + hi * 8;

  for (int t = 0; t < S_LEN; ++t) {
    const size_t pbase = ((size_t)t * BATCH) * HDIM;
    // ---------- phase A: Ar = h@Wr^T, Az = h@Wz^T ----------
    // prefetch this step's pre-activation row (consumed in the epilogue)
    __builtin_prefetch(PA + pbase + (size_t)mbase * HDIM + n_g, 0, 0);
    v8f acc = {0.f,0.f,0.f,0.f,0.f,0.f,0.f,0.f};
    {
      // ping-pong A fragments (global h) against LDS B fragments
      v8bf a0l = *(const v8bf*)(arowA + khiA);
      v8bf a0h = *(const v8bf*)(arowA + khiA + 8);
      for (int kb = 0; kb < 32; kb += 2) {
        const __bf16* p1 = arowA + (kb + 1) * 32 + khiA;
        v8bf a1l = *(const v8bf*)p1, a1h = *(const v8bf*)(p1 + 8);
        v16bf b0 = *(const v16bf*)(browA + kb * 32 + khiB);          // ds_load
        acc = wmma_bf16(pack16(a0l, a0h), b0, acc);
        const int kb2 = (kb + 2 < 32) ? kb + 2 : 31;
        const __bf16* p2 = arowA + kb2 * 32 + khiA;
        a0l = *(const v8bf*)p2; a0h = *(const v8bf*)(p2 + 8);
        v16bf b1 = *(const v16bf*)(browA + (kb + 1) * 32 + khiB);
        acc = wmma_bf16(pack16(a1l, a1h), b1, acc);
      }
    }
    {
      const float bb = bhA[n_g];
#pragma unroll
      for (int v = 0; v < 8; ++v) {
        int m = mbase + v;
        float pre  = acc[v] + PA[pbase + (size_t)m * HDIM + n_g] + bb;
        float gate = 1.f / (1.f + __expf(-pre));
        if (gA == 0) {                                   // r gate -> publish r*h (bf16)
          float rh = gate * hsl[m * 32 + n_l];
          rhbf[(size_t)m * HDIM + n_g] = (__bf16)rh;
        } else {                                         // z gate -> stage locally
          zsl[m * 32 + n_l] = gate;
        }
      }
    }
    gbar(bar, &target);

    // ---------- phase B: An = (r*h)@Wn^T, h' = (1-z)n + z h ----------
    if (wave < 4) {
      __builtin_prefetch(Pn + pbase + (size_t)mbase * HDIM + n_g, 0, 0);
      v8f accn = {0.f,0.f,0.f,0.f,0.f,0.f,0.f,0.f};
      {
        v8bf a0l = *(const v8bf*)(arowB + khiA);
        v8bf a0h = *(const v8bf*)(arowB + khiA + 8);
        for (int kb = 0; kb < 32; kb += 2) {
          const __bf16* p1 = arowB + (kb + 1) * 32 + khiA;
          v8bf a1l = *(const v8bf*)p1, a1h = *(const v8bf*)(p1 + 8);
          v16bf b0 = *(const v16bf*)(browB + kb * 32 + khiB);
          accn = wmma_bf16(pack16(a0l, a0h), b0, accn);
          const int kb2 = (kb + 2 < 32) ? kb + 2 : 31;
          const __bf16* p2 = arowB + kb2 * 32 + khiA;
          a0l = *(const v8bf*)p2; a0h = *(const v8bf*)(p2 + 8);
          v16bf b1 = *(const v16bf*)(browB + (kb + 1) * 32 + khiB);
          accn = wmma_bf16(pack16(a1l, a1h), b1, accn);
        }
      }
      const float bb = bhn[n_g];
#pragma unroll
      for (int v = 0; v < 8; ++v) {
        int m = mbase + v;
        float pre   = accn[v] + Pn[pbase + (size_t)m * HDIM + n_g] + bb;
        float nval  = tanhf(pre);
        float z     = zsl[m * 32 + n_l];
        float hprev = hsl[m * 32 + n_l];
        float hnew  = (1.f - z) * nval + z * hprev;
        hsl[m * 32 + n_l] = hnew;
        hbf[(size_t)m * HDIM + n_g] = (__bf16)hnew;
        if (out_seq)   out_seq  [pbase + (size_t)m * HDIM + n_g] = hnew;
        if (outbf_seq) outbf_seq[pbase + (size_t)m * HDIM + n_g] = (__bf16)hnew;
        if (t == S_LEN - 1) hn_out[(size_t)m * HDIM + n_g] = hnew;
      }
    }
    gbar(bar, &target);
  }
}

// ---------------- host ----------------
static void launch_cvt(const float* s, __bf16* d, size_t n, hipStream_t st) {
  int blocks = (int)((n + 255) / 256);
  if (blocks > 2048) blocks = 2048;
  gru_cvt_bf16<<<blocks, 256, 0, st>>>(s, d, n);
}

extern "C" void kernel_launch(void* const* d_in, const int* in_sizes, int n_in,
                              void* d_out, int out_size, void* d_ws, size_t ws_size,
                              hipStream_t stream) {
  (void)in_sizes; (void)n_in; (void)out_size; (void)ws_size;

  const float* x     = (const float*)d_in[0];
  const float* Wih_r = (const float*)d_in[1];
  const float* Wih_z = (const float*)d_in[2];
  const float* Wih_n = (const float*)d_in[3];
  const float* Whh_r = (const float*)d_in[4];
  const float* Whh_z = (const float*)d_in[5];
  const float* Whh_n = (const float*)d_in[6];
  const float* bih_r = (const float*)d_in[7];
  const float* bih_z = (const float*)d_in[8];
  const float* bih_n = (const float*)d_in[9];
  const float* bhh_r = (const float*)d_in[10];
  const float* bhh_z = (const float*)d_in[11];
  const float* bhh_n = (const float*)d_in[12];
  float* out = (float*)d_out;

  // workspace carve
  char* ws = (char*)d_ws; size_t off = 0;
  auto carve = [&](size_t bytes) -> char* {
    char* p = ws + off; off = (off + bytes + 255) & ~(size_t)255; return p;
  };
  const size_t WIH_E = (size_t)NLAYER * HDIM * FDIM;
  const size_t WHH_E = (size_t)NLAYER * HDIM * HDIM;
  const size_t SEQ_E = (size_t)S_LEN * BATCH * HDIM;

  unsigned* bar   = (unsigned*)carve(256);
  __bf16* WihRb   = (__bf16*)carve(WIH_E * 2);
  __bf16* WihZb   = (__bf16*)carve(WIH_E * 2);
  __bf16* WihNb   = (__bf16*)carve(WIH_E * 2);
  __bf16* WhhRb   = (__bf16*)carve(WHH_E * 2);
  __bf16* WhhZb   = (__bf16*)carve(WHH_E * 2);
  __bf16* WhhNb   = (__bf16*)carve(WHH_E * 2);
  __bf16* Xbf0    = (__bf16*)carve((size_t)S_LEN * BATCH * FDIM * 2);
  __bf16* Xbf1    = (__bf16*)carve(SEQ_E * 2);
  float*  Pr      = (float*)carve(SEQ_E * 4);
  float*  Pz      = (float*)carve(SEQ_E * 4);
  float*  Pn      = (float*)carve(SEQ_E * 4);
  __bf16* hbf     = (__bf16*)carve((size_t)BATCH * HDIM * 2);
  __bf16* rhbf    = (__bf16*)carve((size_t)BATCH * HDIM * 2);

  // allow >64KB dynamic LDS (WGP has 320KB on gfx1250); idempotent, capture-safe
  (void)hipFuncSetAttribute((const void*)gru_recurrent,
                            hipFuncAttributeMaxDynamicSharedMemorySize, REC_LDS_BYTES);

  // bf16 conversions (weights + layer-0 input)
  launch_cvt(x,     Xbf0,  (size_t)S_LEN * BATCH * FDIM, stream);
  launch_cvt(Wih_r, WihRb, WIH_E, stream);
  launch_cvt(Wih_z, WihZb, WIH_E, stream);
  launch_cvt(Wih_n, WihNb, WIH_E, stream);
  launch_cvt(Whh_r, WhhRb, WHH_E, stream);
  launch_cvt(Whh_z, WhhZb, WHH_E, stream);
  launch_cvt(Whh_n, WhhNb, WHH_E, stream);

  const dim3 pgrid(8, 128, 3), pblk(256);
  float* hn_base = out + SEQ_E;   // d_out = [S,B,H] out  ++  [L,B,H] h_n

  // ---- layer 0 ----
  gru_proj_gemm<<<pgrid, pblk, 0, stream>>>(Xbf0, WihRb, WihZb, WihNb,
                                            bih_r, bih_z, bih_n, Pr, Pz, Pn);
  (void)hipMemsetAsync(bar, 0, 256, stream);
  gru_recurrent<<<NWG_REC, REC_THREADS, REC_LDS_BYTES, stream>>>(
      Pr, Pz, Pn, bhh_r, bhh_z, bhh_n, WhhRb, WhhZb, WhhNb,
      hbf, rhbf, /*out_seq=*/nullptr, /*outbf_seq=*/Xbf1,
      /*hn_out=*/hn_base, bar);

  // ---- layer 1 ----
  const size_t wih1 = (size_t)HDIM * FDIM, whh1 = (size_t)HDIM * HDIM;
  gru_proj_gemm<<<pgrid, pblk, 0, stream>>>(Xbf1, WihRb + wih1, WihZb + wih1, WihNb + wih1,
                                            bih_r + HDIM, bih_z + HDIM, bih_n + HDIM,
                                            Pr, Pz, Pn);
  (void)hipMemsetAsync(bar, 0, 256, stream);
  gru_recurrent<<<NWG_REC, REC_THREADS, REC_LDS_BYTES, stream>>>(
      Pr, Pz, Pn, bhh_r + HDIM, bhh_z + HDIM, bhh_n + HDIM,
      WhhRb + whh1, WhhZb + whh1, WhhNb + whh1,
      hbf, rhbf, /*out_seq=*/out, /*outbf_seq=*/nullptr,
      /*hn_out=*/hn_base + (size_t)BATCH * HDIM, bar);
}
